// MentalPlus_58737972740342
// MI455X (gfx1250) — matrix operand order repaired and checked by reading, not compiled, not run on hardware
//
#include <hip/hip_runtime.h>
#include <hip/hip_bf16.h>

typedef __attribute__((ext_vector_type(16))) __bf16 v16bf;
typedef __attribute__((ext_vector_type(8)))  __bf16 v8bf;
typedef __attribute__((ext_vector_type(4)))  __bf16 v4bf;
typedef __attribute__((ext_vector_type(8)))  float  v8f;

#define Bn 8
#define Cn 64
#define Hn 64
#define Wn 64
#define HWn (Hn * Wn)

// gfx1250 async global->LDS copy (ASYNCcnt-tracked), GV mode (saddr=off).
// lds_addr = low 32 bits of generic shared pointer = LDS byte offset.
__device__ __forceinline__ void async_ld_b128(unsigned lds_addr, const float* gaddr) {
    asm volatile("global_load_async_to_lds_b128 %0, %1, off"
                 :: "v"(lds_addr), "v"(gaddr) : "memory");
}
__device__ __forceinline__ void wait_asynccnt0() {
    asm volatile("s_wait_asynccnt 0x0" ::: "memory");
}

// ---------------------------------------------------------------------------
// Phase 1: q/k/v = W{q,k,v} @ x  via v_wmma_f32_16x16x32_bf16.
// x tile stored TRANSPOSED in LDS (xs[s][c]) so every WMMA fragment is two
// contiguous 16B runs -> ds_load_b128 instead of 16x ds_load_u16.
// ---------------------------------------------------------------------------
__global__ void __launch_bounds__(256)
qkv_wmma_kernel(const float* __restrict__ x,
                const float* __restrict__ Wq,
                const float* __restrict__ Wk,
                const float* __restrict__ Wv,
                float* __restrict__ q, float* __restrict__ k, float* __restrict__ v) {
    __shared__ __align__(16) __bf16 xs[128][Cn];     // x[s][c], 16 KB
    __shared__ __align__(16) __bf16 wsm[3][Cn][Cn];  // W[m][o][c], 24 KB

    const int blk   = blockIdx.x;            // B * 32 blocks
    const int b     = blk >> 5;
    const int sbase = (blk & 31) * 128;
    const int tid   = threadIdx.x;

    // Stage x tile (transposed) with b128 global loads.
    const float* xb = x + (size_t)b * Cn * HWn;
    for (int i = tid * 4; i < Cn * 128; i += 1024) {
        const int c = i >> 7, s = i & 127;           // 4 consecutive s, same c
        const float4 xv = *(const float4*)&xb[c * HWn + sbase + s];
        xs[s + 0][c] = (__bf16)xv.x;
        xs[s + 1][c] = (__bf16)xv.y;
        xs[s + 2][c] = (__bf16)xv.z;
        xs[s + 3][c] = (__bf16)xv.w;
    }
    // Stage weights with b128 loads / b64 LDS stores.
    __bf16* w0 = &wsm[0][0][0];
    __bf16* w1 = &wsm[1][0][0];
    __bf16* w2 = &wsm[2][0][0];
    for (int i = tid * 4; i < Cn * Cn; i += 1024) {
        const float4 aq = *(const float4*)&Wq[i];
        const float4 ak = *(const float4*)&Wk[i];
        const float4 av = *(const float4*)&Wv[i];
        *(v4bf*)(w0 + i) = (v4bf){(__bf16)aq.x, (__bf16)aq.y, (__bf16)aq.z, (__bf16)aq.w};
        *(v4bf*)(w1 + i) = (v4bf){(__bf16)ak.x, (__bf16)ak.y, (__bf16)ak.z, (__bf16)ak.w};
        *(v4bf*)(w2 + i) = (v4bf){(__bf16)av.x, (__bf16)av.y, (__bf16)av.z, (__bf16)av.w};
    }
    __syncthreads();

    const int wave  = tid >> 5;   // 0..7 -> N-tile
    const int lane  = tid & 31;
    const int nbase = wave * 16;
    const int ncol  = nbase + (lane & 15);

    // B fragments (32x16): lane holds 16 consecutive K (=c) of its column.
    const v8bf* xrow = (const v8bf*)&xs[ncol][0];
    v16bf bfrag[2];
#pragma unroll
    for (int kc = 0; kc < 2; ++kc) {
        const int kb = kc * 32 + ((lane < 16) ? 0 : 16);   // halves; mult of 8
        const v8bf lo = xrow[(kb >> 3) + 0];
        const v8bf hi = xrow[(kb >> 3) + 1];
        bfrag[kc] = __builtin_shufflevector(lo, hi, 0, 1, 2, 3, 4, 5, 6, 7,
                                            8, 9, 10, 11, 12, 13, 14, 15);
    }

    float* const outs[3] = {q + (size_t)b * Cn * HWn,
                            k + (size_t)b * Cn * HWn,
                            v + (size_t)b * Cn * HWn};

#pragma unroll
    for (int m = 0; m < 3; ++m) {
#pragma unroll
        for (int ob = 0; ob < 4; ++ob) {
            const int arow = ob * 16 + (lane & 15);
            const v8bf* wrow = (const v8bf*)&wsm[m][arow][0];
            v8f acc = {};
#pragma unroll
            for (int kc = 0; kc < 2; ++kc) {
                // A (16x32 bf16): halves 0..15 = K kb..kb+15
                const int kb  = kc * 32 + ((lane < 16) ? 0 : 16);  // mult of 8
                const int idx = kb >> 3;
                const v8bf lo = wrow[idx];
                const v8bf hi = wrow[idx + 1];
                const v16bf a = __builtin_shufflevector(lo, hi, 0, 1, 2, 3, 4, 5, 6, 7,
                                                        8, 9, 10, 11, 12, 13, 14, 15);
                acc = __builtin_amdgcn_wmma_f32_16x16x32_bf16(
                          false, a, false, bfrag[kc], (short)0, acc, false, false);
            }
            // D layout: VGPR r -> M = r + (lane<16 ? 0 : 8); N = lane&15
            const int orow = ob * 16 + ((lane < 16) ? 0 : 8);
            const int scol = sbase + nbase + (lane & 15);
            float* out = outs[m];
#pragma unroll
            for (int r = 0; r < 8; ++r)
                out[(orow + r) * HWn + scol] = acc[r];
        }
    }
}

// ---------------------------------------------------------------------------
// Phase 2: per-channel 5x5-window softmax attention.
// k/v rows staged into LDS with GLOBAL_LOAD_ASYNC_TO_LDS_B128 (ASYNCcnt).
// ---------------------------------------------------------------------------
__global__ void __launch_bounds__(256)
attn_kernel(const float* __restrict__ q, const float* __restrict__ k,
            const float* __restrict__ v, const float* __restrict__ rel_h,
            const float* __restrict__ rel_w, float* __restrict__ out) {
    __shared__ __align__(16) float ks[5][16][Wn];   // 20 KB
    __shared__ __align__(16) float vs[5][16][Wn];   // 20 KB
    __shared__ float bias[16][25];                  // 1.6 KB

    const int bh    = blockIdx.x >> 2;
    const int cbase = (blockIdx.x & 3) * 16;
    const int b     = bh / Hn;
    const int h     = bh % Hn;
    const int tid   = threadIdx.x;

    const float* qb = q + (size_t)b * Cn * HWn;
    __builtin_prefetch(qb + cbase * HWn + h * Wn, 0, 1);   // global_prefetch_b8

    for (int i = tid; i < 16 * 25; i += 256) {
        const int cl = i / 25, j = i % 25;
        const int c  = cbase + cl;
        const int dy = j / 5, dx = j % 5;
        bias[cl][j] = (c < 32) ? rel_h[c * 5 + dy] : rel_w[(c - 32) * 5 + dx];
    }

    const float* kb = k + (size_t)b * Cn * HWn;
    const float* vb = v + (size_t)b * Cn * HWn;
    // 5 rows x 16 ch x 64 w, float4 granularity: 1280 vecs, 5 per thread.
    for (int j = tid; j < 5 * 16 * 16; j += 256) {
        const int flat = j * 4;
        const int dy = flat >> 10;
        const int cl = (flat >> 6) & 15;
        const int w  = flat & 63;                    // multiple of 4
        const int hh = h + dy - 2;
        float* kd = &ks[dy][cl][w];
        float* vd = &vs[dy][cl][w];
        if (hh >= 0 && hh < Hn) {
            const int gi = (cbase + cl) * HWn + hh * Wn + w;
            async_ld_b128((unsigned)(size_t)kd, kb + gi);
            async_ld_b128((unsigned)(size_t)vd, vb + gi);
        } else {
            *(float4*)kd = (float4){0.f, 0.f, 0.f, 0.f};
            *(float4*)vd = (float4){0.f, 0.f, 0.f, 0.f};
        }
    }
    wait_asynccnt0();
    __syncthreads();

    float* ob = out + (size_t)b * Cn * HWn;

    for (int i = tid; i < 16 * Wn; i += 256) {
        const int cl = i >> 6;
        const int w  = i & 63;
        const float qv = qb[(cbase + cl) * HWn + h * Wn + w];

        float mx = -1e30f;
#pragma unroll
        for (int dy = 0; dy < 5; ++dy) {
#pragma unroll
            for (int dx = 0; dx < 5; ++dx) {
                const int ww = w + dx - 2;
                const float kk = (ww >= 0 && ww < Wn) ? ks[dy][cl][ww] : 0.0f;
                const float lg = qv * (kk + bias[cl][dy * 5 + dx]);
                mx = fmaxf(mx, lg);
            }
        }
        float se = 0.0f, sv = 0.0f;
#pragma unroll
        for (int dy = 0; dy < 5; ++dy) {
#pragma unroll
            for (int dx = 0; dx < 5; ++dx) {
                const int ww = w + dx - 2;
                const bool in = (ww >= 0 && ww < Wn);
                const float kk = in ? ks[dy][cl][ww] : 0.0f;
                const float vv = in ? vs[dy][cl][ww] : 0.0f;
                const float e  = __expf(qv * (kk + bias[cl][dy * 5 + dx]) - mx);
                se += e;
                sv += e * vv;
            }
        }
        ob[(cbase + cl) * HWn + h * Wn + w] = sv / se;
    }
}

extern "C" void kernel_launch(void* const* d_in, const int* in_sizes, int n_in,
                              void* d_out, int out_size, void* d_ws, size_t ws_size,
                              hipStream_t stream) {
    const float* x     = (const float*)d_in[0];
    const float* Wq    = (const float*)d_in[1];
    const float* Wk    = (const float*)d_in[2];
    const float* Wv    = (const float*)d_in[3];
    const float* rel_h = (const float*)d_in[4];
    const float* rel_w = (const float*)d_in[5];
    float* out = (float*)d_out;

    const size_t plane = (size_t)Bn * Cn * HWn;   // 2,097,152 floats = 8 MB
    float* q = (float*)d_ws;
    float* k = q + plane;
    float* v = k + plane;

    qkv_wmma_kernel<<<Bn * 32, 256, 0, stream>>>(x, Wq, Wk, Wv, q, k, v);
    attn_kernel<<<Bn * Hn * 4, 256, 0, stream>>>(q, k, v, rel_h, rel_w, out);
}
